// RFDAT_10806137716845
// MI455X (gfx1250) — matrix operand order repaired and checked by reading, not compiled, not run on hardware
//
#include <hip/hip_runtime.h>
#include <math.h>

#define N_USERC 100000
#define N_ITEMC 50000
#define N_TOTALC (N_USERC + N_ITEMC)
#define DD 64
#define NNZ_AC 2000000
#define NNZ_SC 1000000
#define N_TAILC 10000
#define N_TAIL_PAD 10016   // 313 * 32
#define N_HEADC 4000
#define TOPKC 10
#define BATCHC 4096
#define NNEGC 4
#define CONVF 40.0f

typedef __attribute__((ext_vector_type(16))) __bf16 v16bf;
typedef __attribute__((ext_vector_type(8)))  float  v8f;

struct bfpack { uint4 lo, hi; };   // 32 bytes -> bit_cast to v16bf

// ---------------- utility kernels ----------------

__global__ void k_zero_f32(float* __restrict__ p, long n) {
  long i = (long)blockIdx.x * blockDim.x + threadIdx.x;
  long st = (long)gridDim.x * blockDim.x;
  for (; i < n; i += st) p[i] = 0.0f;
}

__global__ void k_copy_f32(float* __restrict__ dst, const float* __restrict__ src, long n) {
  long i = (long)blockIdx.x * blockDim.x + threadIdx.x;
  long st = (long)gridDim.x * blockDim.x;
  for (; i < n; i += st) dst[i] = src[i];
}

__global__ void k_add_f32(float* __restrict__ dst, const float* __restrict__ src, long n) {
  long i = (long)blockIdx.x * blockDim.x + threadIdx.x;
  long st = (long)gridDim.x * blockDim.x;
  for (; i < n; i += st) dst[i] += src[i];
}

__global__ void k_scale_f32(float* __restrict__ dst, float s, long n) {
  long i = (long)blockIdx.x * blockDim.x + threadIdx.x;
  long st = (long)gridDim.x * blockDim.x;
  for (; i < n; i += st) dst[i] *= s;
}

// ---------------- fold the (linear) 2-layer MLP into M(64x64), c(64) ----------------
// enc(f) = f @ M + c, M[d][j] = sum_k w0[k][d]*w2[j][k], c[j] = sum_k b0[k]*w2[j][k] + b2[j]
__global__ void k_fold_mlp(const float* __restrict__ w0, const float* __restrict__ b0,
                           const float* __restrict__ w2, const float* __restrict__ b2,
                           float* __restrict__ M, float* __restrict__ c) {
  int g = blockIdx.x * blockDim.x + threadIdx.x;
  if (g >= 64 * 64) return;
  int j = g & 63, d = g >> 6;
  float acc = 0.0f;
  for (int k = 0; k < 256; ++k) acc += w0[k * 64 + d] * w2[j * 256 + k];
  M[d * 64 + j] = acc;
  if (d == 0) {
    float cc = b2[j];
    for (int k = 0; k < 256; ++k) cc += b0[k] * w2[j * 256 + k];
    c[j] = cc;
  }
}

// ---------------- encode gathered rows to bf16: out[r] = emb[idx[r]] @ M + c ----------------
__global__ void k_encode_bf16(const float* __restrict__ emb, const int* __restrict__ idx, int n,
                              const float* __restrict__ M, const float* __restrict__ c,
                              __bf16* __restrict__ out) {
  int r = blockIdx.x;
  if (r >= n) return;
  int t = threadIdx.x;  // 64 threads
  __shared__ float f[64];
  f[t] = emb[(long)idx[r] * DD + t];
  __syncthreads();
  float acc = c[t];
  #pragma unroll 8
  for (int d = 0; d < 64; ++d) acc += f[d] * M[d * 64 + t];
  out[(long)r * DD + t] = (__bf16)acc;
}

// ---------------- fused scores GEMM (bf16 WMMA) + streaming top-K + weight emit ----------------
// One wave per 32 tail rows (two 16x16 A tiles). 4x v_wmma_f32_16x16x32_bf16 per head tile,
// B loaded once per head tile and shared by both A tiles. All 32 lanes own one row for top-K.
__global__ __launch_bounds__(32)
void k_scores_topk(const __bf16* __restrict__ tailE, const __bf16* __restrict__ headE,
                   const int* __restrict__ head_idx,
                   float* __restrict__ igw, int* __restrict__ igcol) {
  const int lane  = threadIdx.x;
  const int lrow  = lane & 15;
  const int khalf = (lane >> 4) * 8;   // A-matrix per-lane K split (ISA 16-bit A layout)
  const int kb    = (lane >> 4) * 8;   // B-matrix per-lane K split (lane = N, same K order as A)
  const long base = (long)blockIdx.x * 32;

  // A tiles: rows [base, base+16) and [base+16, base+32), K=64 as two K=32 chunks.
  const __bf16* ar0 = tailE + (base + lrow) * DD;
  const __bf16* ar1 = tailE + (base + 16 + lrow) * DD;
  bfpack p;
  p.lo = *(const uint4*)(ar0 + khalf);      p.hi = *(const uint4*)(ar0 + 16 + khalf);
  v16bf a00 = __builtin_bit_cast(v16bf, p);
  p.lo = *(const uint4*)(ar0 + 32 + khalf); p.hi = *(const uint4*)(ar0 + 48 + khalf);
  v16bf a01 = __builtin_bit_cast(v16bf, p);
  p.lo = *(const uint4*)(ar1 + khalf);      p.hi = *(const uint4*)(ar1 + 16 + khalf);
  v16bf a10 = __builtin_bit_cast(v16bf, p);
  p.lo = *(const uint4*)(ar1 + 32 + khalf); p.hi = *(const uint4*)(ar1 + 48 + khalf);
  v16bf a11 = __builtin_bit_cast(v16bf, p);

  float tv[TOPKC]; int ti[TOPKC];
  #pragma unroll
  for (int k = 0; k < TOPKC; ++k) { tv[k] = -3.0e38f; ti[k] = 0; }

  __shared__ float sc[2 * 16 * 16];
  const int mbase = (lane >> 4) * 8;
  float* myrow = sc + (lane >> 4) * 256 + lrow * 16;  // row this lane owns in the scan

  for (int ht = 0; ht < N_HEADC / 16; ++ht) {
    const __bf16* brow = headE + (long)(ht * 16 + lrow) * DD;
    bfpack q;
    q.lo = *(const uint4*)(brow + kb);          q.hi = *(const uint4*)(brow + 16 + kb);
    v16bf b0v = __builtin_bit_cast(v16bf, q);
    q.lo = *(const uint4*)(brow + 32 + kb);     q.hi = *(const uint4*)(brow + 48 + kb);
    v16bf b1v = __builtin_bit_cast(v16bf, q);

    v8f c0 = {}, c1 = {};
    c0 = __builtin_amdgcn_wmma_f32_16x16x32_bf16(false, a00, false, b0v, (short)0, c0, false, false);
    c1 = __builtin_amdgcn_wmma_f32_16x16x32_bf16(false, a10, false, b0v, (short)0, c1, false, false);
    c0 = __builtin_amdgcn_wmma_f32_16x16x32_bf16(false, a01, false, b1v, (short)0, c0, false, false);
    c1 = __builtin_amdgcn_wmma_f32_16x16x32_bf16(false, a11, false, b1v, (short)0, c1, false, false);

    // C/D layout: lanes 0-15 -> M=v, lanes 16-31 -> M=v+8; N = lane&15
    #pragma unroll
    for (int v = 0; v < 8; ++v) {
      sc[(mbase + v) * 16 + lrow]       = c0[v];
      sc[256 + (mbase + v) * 16 + lrow] = c1[v];
    }
    __syncthreads();

    // every lane owns one of the 32 rows: tile = lane>>4, row-in-tile = lane&15
    #pragma unroll
    for (int n = 0; n < 16; ++n) {
      float v = myrow[n];
      if (v > tv[TOPKC - 1]) {
        tv[TOPKC - 1] = v; ti[TOPKC - 1] = ht * 16 + n;
        #pragma unroll
        for (int k = TOPKC - 1; k >= 1; --k) {
          if (tv[k] > tv[k - 1]) {
            float tf = tv[k]; tv[k] = tv[k - 1]; tv[k - 1] = tf;
            int   tt = ti[k]; ti[k] = ti[k - 1]; ti[k - 1] = tt;
          }
        }
      }
    }
    __syncthreads();
  }

  long r = base + lane;  // (lane>>4)*16 + (lane&15) == lane
  if (r < N_TAILC) {
    float s[TOPKC]; float ssum = 0.0f;
    #pragma unroll
    for (int k = 0; k < TOPKC; ++k) { s[k] = 1.0f / (1.0f + expf(-tv[k])); ssum += s[k]; }
    float inv = 1.0f / (ssum + 1.0f);
    #pragma unroll
    for (int k = 0; k < TOPKC; ++k) {
      igw[r * TOPKC + k]   = s[k] * inv;
      igcol[r * TOPKC + k] = head_idx[ti[k]];
    }
  }
}

// ---------------- SpMM scatter: y[rows[i]] += vals[i] * x[cols[i]] (64 lanes / nnz) ----------------
__global__ void k_spmm_scatter(const float* __restrict__ vals, const int* __restrict__ rows,
                               const int* __restrict__ cols, const float* __restrict__ x,
                               float* __restrict__ y, long nnz) {
  long g = (long)blockIdx.x * blockDim.x + threadIdx.x;
  long st = (long)gridDim.x * blockDim.x;
  long total = nnz * DD;
  for (; g < total; g += st) {
    long i = g >> 6; int d = (int)(g & 63);
    atomicAdd(&y[(long)rows[i] * DD + d], vals[i] * x[(long)cols[i] * DD + d]);
  }
}

// ---------------- item-graph scatter: enh[tail_idx[e/K]] += igw[e] * items[igcol[e]] ----------------
__global__ void k_enh_scatter(const float* __restrict__ igw, const int* __restrict__ igcol,
                              const int* __restrict__ tail_idx, const float* __restrict__ items,
                              float* __restrict__ enh) {
  long g = (long)blockIdx.x * blockDim.x + threadIdx.x;
  long st = (long)gridDim.x * blockDim.x;
  long total = (long)N_TAILC * TOPKC * DD;
  for (; g < total; g += st) {
    long e = g >> 6; int d = (int)(g & 63);
    int r = (int)(e / TOPKC);
    atomicAdd(&enh[(long)tail_idx[r] * DD + d], igw[e] * items[(long)igcol[e] * DD + d]);
  }
}

// ---------------- items_final = items + sum_weight[:,None] * enh ----------------
__global__ void k_finalize_items(const float* __restrict__ items, const float* __restrict__ enh,
                                 const float* __restrict__ deg, float* __restrict__ out) {
  long g = (long)blockIdx.x * blockDim.x + threadIdx.x;
  long st = (long)gridDim.x * blockDim.x;
  long total = (long)N_ITEMC * DD;
  for (; g < total; g += st) {
    long i = g >> 6;
    float sw = CONVF / (CONVF + expf(deg[i] / CONVF));
    out[g] = items[g] + sw * enh[g];
  }
}

// ---------------- BPR loss + reg ----------------
__global__ void k_loss(const float* __restrict__ uf, const float* __restrict__ itf,
                       const float* __restrict__ user_emb, const float* __restrict__ item_emb,
                       const int* __restrict__ bu, const int* __restrict__ bp,
                       const int* __restrict__ bn, float* __restrict__ out) {
  int b = blockIdx.x * blockDim.x + threadIdx.x;
  if (b >= BATCHC) return;
  const float* ue = uf + (long)bu[b] * DD;
  const float* pe = itf + (long)bp[b] * DD;
  const float* u0 = user_emb + (long)bu[b] * DD;
  const float* p0 = item_emb + (long)bp[b] * DD;
  float pos = 0.0f, reg = 0.0f;
  #pragma unroll 4
  for (int d = 0; d < DD; ++d) { pos += ue[d] * pe[d]; reg += u0[d] * u0[d] + p0[d] * p0[d]; }
  float neg = 0.0f;
  for (int j = 0; j < NNEGC; ++j) {
    int ni = bn[b * NNEGC + j];
    const float* ne = itf + (long)ni * DD;
    const float* n0 = item_emb + (long)ni * DD;
    float nd = 0.0f;
    #pragma unroll 4
    for (int d = 0; d < DD; ++d) { nd += ue[d] * ne[d]; reg += n0[d] * n0[d]; }
    neg += nd;
  }
  neg *= (1.0f / NNEGC);
  float x = neg - pos;
  float sp = (x > 20.0f) ? x : log1pf(expf(x));
  atomicAdd(&out[0], sp / (float)BATCHC);
  atomicAdd(&out[1], 0.5f * reg / (float)BATCHC);
}

// ---------------- host launch ----------------

static inline unsigned gsz(long n) {
  long b = (n + 255) / 256;
  if (b > 1048576) b = 1048576;
  if (b < 1) b = 1;
  return (unsigned)b;
}

extern "C" void kernel_launch(void* const* d_in, const int* in_sizes, int n_in,
                              void* d_out, int out_size, void* d_ws, size_t ws_size,
                              hipStream_t stream) {
  (void)in_sizes; (void)n_in; (void)out_size; (void)ws_size;
  const float* user_emb    = (const float*)d_in[0];
  const float* item_emb    = (const float*)d_in[1];
  const float* w0          = (const float*)d_in[2];
  const float* b0          = (const float*)d_in[3];
  const float* w2          = (const float*)d_in[4];
  const float* b2          = (const float*)d_in[5];
  const float* a_vals      = (const float*)d_in[6];
  const float* s_vals      = (const float*)d_in[7];
  const float* item_degree = (const float*)d_in[8];
  const int*   a_rows      = (const int*)d_in[9];
  const int*   a_cols      = (const int*)d_in[10];
  const int*   s_rows      = (const int*)d_in[11];
  const int*   s_cols      = (const int*)d_in[12];
  const int*   tail_idx    = (const int*)d_in[13];
  const int*   head_idx    = (const int*)d_in[14];
  const int*   batch_user  = (const int*)d_in[15];
  const int*   batch_pos   = (const int*)d_in[16];
  const int*   batch_neg   = (const int*)d_in[17];
  float* out = (float*)d_out;

  char* ws = (char*)d_ws;
  size_t off = 0;
  #define TAKE(name, type, count) \
    type* name = (type*)(ws + off); \
    off = (off + (size_t)(count) * sizeof(type) + 255) & ~(size_t)255;

  TAKE(fM,    float,  64 * 64)
  TAKE(fc,    float,  64)
  TAKE(tailE, __bf16, (size_t)N_TAIL_PAD * DD)
  TAKE(headE, __bf16, (size_t)N_HEADC * DD)
  TAKE(igw,   float,  (size_t)N_TAILC * TOPKC)
  TAKE(igcol, int,    (size_t)N_TAILC * TOPKC)
  TAKE(E0,    float,  (size_t)N_TOTALC * DD)
  TAKE(E1,    float,  (size_t)N_TOTALC * DD)
  TAKE(SA,    float,  (size_t)N_USERC * DD)
  TAKE(SB,    float,  (size_t)N_USERC * DD)
  TAKE(ACC,   float,  (size_t)N_USERC * DD)
  TAKE(ENH,   float,  (size_t)N_ITEMC * DD)
  TAKE(IFIN,  float,  (size_t)N_ITEMC * DD)
  #undef TAKE

  // output accumulators start at 0
  k_zero_f32<<<1, 32, 0, stream>>>(out, 2);

  // fold linear encoder, encode tail/head rows to bf16
  k_fold_mlp<<<16, 256, 0, stream>>>(w0, b0, w2, b2, fM, fc);
  k_encode_bf16<<<N_TAILC, 64, 0, stream>>>(item_emb, tail_idx, N_TAILC, fM, fc, tailE);
  k_encode_bf16<<<N_HEADC, 64, 0, stream>>>(item_emb, head_idx, N_HEADC, fM, fc, headE);

  // fused WMMA GEMM + top-K -> item-graph weights (32 tail rows per wave)
  k_scores_topk<<<N_TAIL_PAD / 32, 32, 0, stream>>>(tailE, headE, head_idx, igw, igcol);

  // E0 = concat(user_emb, item_emb)
  k_copy_f32<<<gsz((long)N_USERC * DD), 256, 0, stream>>>(E0, user_emb, (long)N_USERC * DD);
  k_copy_f32<<<gsz((long)N_ITEMC * DD), 256, 0, stream>>>(E0 + (size_t)N_USERC * DD, item_emb, (long)N_ITEMC * DD);

  // 3 interaction-graph SpMM layers, ping-pong E0 <-> E1
  float* Ein = E0; float* Eout = E1;
  for (int l = 0; l < 3; ++l) {
    k_zero_f32<<<gsz((long)N_TOTALC * DD), 256, 0, stream>>>(Eout, (long)N_TOTALC * DD);
    k_spmm_scatter<<<gsz((long)NNZ_AC * DD), 256, 0, stream>>>(a_vals, a_rows, a_cols, Ein, Eout, (long)NNZ_AC);
    float* t = Ein; Ein = Eout; Eout = t;
  }
  float* Efin  = Ein;                       // result of 3 layers
  float* users = Efin;
  float* items = Efin + (size_t)N_USERC * DD;

  // social smoothing: ACC = users + S@users + S@S@users, then /3
  k_copy_f32<<<gsz((long)N_USERC * DD), 256, 0, stream>>>(ACC, users, (long)N_USERC * DD);
  k_zero_f32<<<gsz((long)N_USERC * DD), 256, 0, stream>>>(SA, (long)N_USERC * DD);
  k_spmm_scatter<<<gsz((long)NNZ_SC * DD), 256, 0, stream>>>(s_vals, s_rows, s_cols, users, SA, (long)NNZ_SC);
  k_add_f32<<<gsz((long)N_USERC * DD), 256, 0, stream>>>(ACC, SA, (long)N_USERC * DD);
  k_zero_f32<<<gsz((long)N_USERC * DD), 256, 0, stream>>>(SB, (long)N_USERC * DD);
  k_spmm_scatter<<<gsz((long)NNZ_SC * DD), 256, 0, stream>>>(s_vals, s_rows, s_cols, SA, SB, (long)NNZ_SC);
  k_add_f32<<<gsz((long)N_USERC * DD), 256, 0, stream>>>(ACC, SB, (long)N_USERC * DD);
  k_scale_f32<<<gsz((long)N_USERC * DD), 256, 0, stream>>>(ACC, 1.0f / 3.0f, (long)N_USERC * DD);

  // item enhancement via top-K graph + degree-gated blend
  k_zero_f32<<<gsz((long)N_ITEMC * DD), 256, 0, stream>>>(ENH, (long)N_ITEMC * DD);
  k_enh_scatter<<<gsz((long)N_TAILC * TOPKC * DD), 256, 0, stream>>>(igw, igcol, tail_idx, items, ENH);
  k_finalize_items<<<gsz((long)N_ITEMC * DD), 256, 0, stream>>>(items, ENH, item_degree, IFIN);

  // BPR loss + regularizer
  k_loss<<<(BATCHC + 255) / 256, 256, 0, stream>>>(ACC, IFIN, user_emb, item_emb,
                                                   batch_user, batch_pos, batch_neg, out);
}
